// GNCAFFamilyB_76948634075446
// MI455X (gfx1250) — compile-verified, matrix operand order checked by reading, not run on hardware
//
#include <hip/hip_runtime.h>
#include <hip/hip_bf16.h>

// ---------------------------------------------------------------------------
// MI455X (gfx1250): all matmul-shaped work -> v_wmma_f32_16x16x32_bf16.
//  - Every GEMM operand is produced as bf16 with K padded to a multiple of 32
//    (zero fill), so the GEMM tile fill is pure b128 loads, no per-element
//    guards (previous round's valu/salu explosion came from those guards).
//  - 128x64 tile / 256 threads (8 waves): each wave owns 16 rows x 64 cols
//    = 4 WMMA tiles per K-step, A fragment reused 4x.
//  - Single operand convention: B is [N, Kpad] row-major (attention V is
//    materialized transposed so A*V also uses it).
// ---------------------------------------------------------------------------

typedef __bf16 bf16_t;
typedef __bf16 v16bf __attribute__((ext_vector_type(16)));
typedef __bf16 v8bf  __attribute__((ext_vector_type(8)));
typedef float  v8f   __attribute__((ext_vector_type(8)));

#define TM 128
#define TN 64
#define TKK 32
#define LDK 40   // +8 bf16 pad: 16B-aligned v8bf LDS accesses, conflict break

static __host__ __device__ inline int align32i(int x) { return (x + 31) & ~31; }

// =====================  WMMA GEMM  =========================================
// C[M,N] = act( A[M,K] * B[N,K]^T + bias ), A/B bf16 with K % 32 == 0,
// rows 16B aligned. Batched via blockIdx.z (element strides).
__global__ __launch_bounds__(256)
void k_gemm_wmma(const bf16_t* __restrict__ A, const bf16_t* __restrict__ B,
                 const float* __restrict__ bias, float* __restrict__ C,
                 int M, int N, int K,
                 long long sA, long long sB, long long sC, int act)
{
    __shared__ bf16_t As[TM][LDK];
    __shared__ bf16_t Bs[TN][LDK];

    A += (long long)blockIdx.z * sA;
    B += (long long)blockIdx.z * sB;
    C += (long long)blockIdx.z * sC;

    const int tid  = threadIdx.x;
    const int lane = tid & 31;
    const int wave = tid >> 5;           // 0..7

    const int tileM = blockIdx.y * TM;
    const int tileN = blockIdx.x * TN;

    const int rowBase = wave * 16;       // wave owns rows [rowBase, +16)
    const int laneRow = lane & 15;
    const int laneHi  = lane >> 4;       // 0|1
    const int kOff    = laneHi * 8;      // ISA 16-bit A/B fragment K groups

    v8f acc0 = {}, acc1 = {}, acc2 = {}, acc3 = {};

    // A-fill coords: 256 threads x 16 elems = 128x32
    const int ar = tid >> 1;
    const int ac = (tid & 1) * 16;
    // B-fill coords: 256 threads x 8 elems = 64x32
    const int br = tid >> 2;
    const int bc = (tid & 3) * 8;

    for (int k0 = 0; k0 < K; k0 += TKK) {
        if (k0 + TKK < K) {
            __builtin_prefetch(A + (long long)(tileM + ar) * K + k0 + TKK, 0, 1);
        }
        {
            const v8bf z = {};
            v8bf v0 = z, v1 = z;
            if (tileM + ar < M) {
                const bf16_t* src = A + (long long)(tileM + ar) * K + k0 + ac;
                v0 = *(const v8bf*)src;
                v1 = *(const v8bf*)(src + 8);
            }
            *(v8bf*)&As[ar][ac]     = v0;
            *(v8bf*)&As[ar][ac + 8] = v1;
        }
        {
            const v8bf z = {};
            v8bf v0 = z;
            if (tileN + br < N)
                v0 = *(const v8bf*)(B + (long long)(tileN + br) * K + k0 + bc);
            *(v8bf*)&Bs[br][bc] = v0;
        }
        __syncthreads();

        union { v16bf v; v8bf h[2]; } fa, fb;
        fa.h[0] = *(const v8bf*)&As[rowBase + laneRow][kOff];
        fa.h[1] = *(const v8bf*)&As[rowBase + laneRow][kOff + 16];

        fb.h[0] = *(const v8bf*)&Bs[laneRow][kOff];
        fb.h[1] = *(const v8bf*)&Bs[laneRow][kOff + 16];
        acc0 = __builtin_amdgcn_wmma_f32_16x16x32_bf16(false, fa.v, false, fb.v,
                                                       (short)0, acc0, false, false);
        fb.h[0] = *(const v8bf*)&Bs[16 + laneRow][kOff];
        fb.h[1] = *(const v8bf*)&Bs[16 + laneRow][kOff + 16];
        acc1 = __builtin_amdgcn_wmma_f32_16x16x32_bf16(false, fa.v, false, fb.v,
                                                       (short)0, acc1, false, false);
        fb.h[0] = *(const v8bf*)&Bs[32 + laneRow][kOff];
        fb.h[1] = *(const v8bf*)&Bs[32 + laneRow][kOff + 16];
        acc2 = __builtin_amdgcn_wmma_f32_16x16x32_bf16(false, fa.v, false, fb.v,
                                                       (short)0, acc2, false, false);
        fb.h[0] = *(const v8bf*)&Bs[48 + laneRow][kOff];
        fb.h[1] = *(const v8bf*)&Bs[48 + laneRow][kOff + 16];
        acc3 = __builtin_amdgcn_wmma_f32_16x16x32_bf16(false, fa.v, false, fb.v,
                                                       (short)0, acc3, false, false);
        __syncthreads();
    }

    // store per ISA 32-bit 16x16 C/D layout (col = lane%16, row = v + 8*hi)
    v8f* accs[4] = { &acc0, &acc1, &acc2, &acc3 };
#pragma unroll
    for (int t = 0; t < 4; ++t) {
        int col = tileN + t * 16 + laneRow;
        if (col >= N) continue;
        float bs = bias ? bias[col] : 0.0f;
        v8f& a = *accs[t];
#pragma unroll
        for (int v = 0; v < 8; ++v) {
            int m = tileM + rowBase + v + laneHi * 8;
            if (m >= M) continue;
            float x = a[v] + bs;
            if (act) x = fmaxf(x, 0.0f);
            C[(long long)m * N + col] = x;
        }
    }
}

// =====================  conversion / layout kernels  =======================
// [rows,K] f32 -> [rows,Kp] bf16, zero padded (Kp % 32 == 0).
__global__ void k_f32_to_bf16_pad(const float* __restrict__ x, bf16_t* __restrict__ y,
                                  long long rows, int K, int Kp) {
    long long total = rows * Kp;
    for (long long i = (long long)blockIdx.x * blockDim.x + threadIdx.x; i < total;
         i += (long long)gridDim.x * blockDim.x) {
        int c = (int)(i % Kp);
        long long r = i / Kp;
        y[i] = (c < K) ? (bf16_t)x[r * K + c] : (bf16_t)0.0f;
    }
}

__global__ void k_nchw2nhwc(const float* __restrict__ x, float* __restrict__ y,
                            int Nb, int C, int H, int W) {
    long long total = (long long)Nb * C * H * W;
    for (long long i = (long long)blockIdx.x * blockDim.x + threadIdx.x; i < total;
         i += (long long)gridDim.x * blockDim.x) {
        int w = (int)(i % W); long long t = i / W;
        int h = (int)(t % H); t /= H;
        int c = (int)(t % C); int n = (int)(t / C);
        y[(((long long)n * H + h) * W + w) * C + c] = x[i];
    }
}

__global__ void k_nhwc2nchw(const float* __restrict__ x, float* __restrict__ y,
                            int Nb, int H, int W, int C) {
    long long total = (long long)Nb * H * W * C;
    for (long long i = (long long)blockIdx.x * blockDim.x + threadIdx.x; i < total;
         i += (long long)gridDim.x * blockDim.x) {
        int c = (int)(i % C); long long t = i / C;
        int w = (int)(t % W); t /= W;
        int h = (int)(t % H); int n = (int)(t / H);
        y[(((long long)n * C + c) * H + h) * W + w] = x[i];
    }
}

__global__ void k_affine_act(const float* __restrict__ x, const float* __restrict__ scale,
                             const float* __restrict__ shift, const float* __restrict__ res,
                             float* __restrict__ y, long long rows, int C, int act) {
    long long total = rows * C;
    for (long long i = (long long)blockIdx.x * blockDim.x + threadIdx.x; i < total;
         i += (long long)gridDim.x * blockDim.x) {
        int c = (int)(i % C);
        float v = x[i];
        if (scale) v *= scale[c];
        if (shift) v += shift[c];
        if (res)   v += res[i];
        if (act)   v = fmaxf(v, 0.0f);
        y[i] = v;
    }
}

__global__ void k_bn_prep(const float* g, const float* b, const float* m,
                          const float* v, float* scale, float* shift, int C) {
    int c = blockIdx.x * blockDim.x + threadIdx.x;
    if (c < C) {
        float s = g[c] * rsqrtf(v[c] + 1e-5f);
        scale[c] = s;
        shift[c] = b[c] - m[c] * s;
    }
}

// NHWC im2col -> [rows, Kp] bf16, zero padded. Kd = C*KH*KW, col order (kh,kw,c).
__global__ void k_im2col_pad(const float* __restrict__ x, bf16_t* __restrict__ col,
                             int Nb, int H, int W, int C, int KH, int KW,
                             int stride, int pad, int OH, int OW, int Kd, int Kp) {
    long long rows = (long long)Nb * OH * OW;
    long long total = rows * Kp;
    for (long long i = (long long)blockIdx.x * blockDim.x + threadIdx.x; i < total;
         i += (long long)gridDim.x * blockDim.x) {
        int kcol = (int)(i % Kp);
        long long pix = i / Kp;
        float v = 0.0f;
        if (kcol < Kd) {
            int cc = kcol % C; int t2 = kcol / C;
            int kw = t2 % KW;  int kh = t2 / KW;
            int ow = (int)(pix % OW); long long t = pix / OW;
            int oh = (int)(t % OH);   int n = (int)(t / OH);
            int ih = oh * stride - pad + kh;
            int iw = ow * stride - pad + kw;
            if (ih >= 0 && ih < H && iw >= 0 && iw < W)
                v = x[(((long long)n * H + ih) * W + iw) * C + cc];
        }
        col[i] = (bf16_t)v;
    }
}

// OIHW f32 -> [O, Kp] bf16 (cols ordered (kh,kw,i)), zero padded.
__global__ void k_pack_w_pad(const float* __restrict__ w, bf16_t* __restrict__ wp,
                             int O, int I, int KH, int KW, int Kd, int Kp) {
    long long total = (long long)O * Kp;
    for (long long i = (long long)blockIdx.x * blockDim.x + threadIdx.x; i < total;
         i += (long long)gridDim.x * blockDim.x) {
        int kcol = (int)(i % Kp);
        int o = (int)(i / Kp);
        bf16_t v = (bf16_t)0.0f;
        if (kcol < Kd) {
            int ci = kcol % I; int t2 = kcol / I;
            int kw = t2 % KW;  int kh = t2 / KW;
            v = (bf16_t)w[(((long long)o * I + ci) * KH + kh) * KW + kw];
        }
        wp[i] = v;
    }
}

// convT 2x2: [Cin][Cout][2][2] f32 -> [(co,kh,kw)][Cinp] bf16 (flipped, padded)
__global__ void k_pack_wt_pad(const float* __restrict__ w, bf16_t* __restrict__ wp,
                              int Cin, int Cout, int Cinp) {
    long long total = (long long)Cout * 4 * Cinp;
    for (long long i = (long long)blockIdx.x * blockDim.x + threadIdx.x; i < total;
         i += (long long)gridDim.x * blockDim.x) {
        int ci = (int)(i % Cinp);
        int row = (int)(i / Cinp);
        int r = row & 3; int co = row >> 2;
        int kh = r >> 1; int kw = r & 1;
        bf16_t v = (bf16_t)0.0f;
        if (ci < Cin)
            v = (bf16_t)w[(((long long)ci * Cout + co) * 2 + (1 - kh)) * 2 + (1 - kw)];
        wp[i] = v;
    }
}

__global__ void k_convt_scatter(const float* __restrict__ yg, const float* __restrict__ bias,
                                float* __restrict__ out, int Nb, int H, int W, int Cout) {
    long long total = (long long)Nb * H * W * Cout * 4;
    for (long long i = (long long)blockIdx.x * blockDim.x + threadIdx.x; i < total;
         i += (long long)gridDim.x * blockDim.x) {
        int col = (int)(i % (Cout * 4));
        long long pix = i / (Cout * 4);
        int co = col >> 2; int r = col & 3; int kh = r >> 1; int kw = r & 1;
        int iw = (int)(pix % W); long long t = pix / W;
        int ih = (int)(t % H);   int n = (int)(t / H);
        out[(((long long)n * 2 * H + (2 * ih + kh)) * (2LL * W) + (2 * iw + kw)) * Cout + co]
            = yg[i] + bias[co];
    }
}

__global__ void k_maxpool(const float* __restrict__ x, float* __restrict__ y,
                          int Nb, int H, int W, int C, int OH, int OW) {
    long long total = (long long)Nb * OH * OW * C;
    for (long long i = (long long)blockIdx.x * blockDim.x + threadIdx.x; i < total;
         i += (long long)gridDim.x * blockDim.x) {
        int c = (int)(i % C); long long t = i / C;
        int ow = (int)(t % OW); t /= OW;
        int oh = (int)(t % OH); int n = (int)(t / OH);
        float m = -3.4e38f;
        for (int kh = 0; kh < 3; ++kh)
            for (int kw = 0; kw < 3; ++kw) {
                int ih = oh * 2 - 1 + kh, iw = ow * 2 - 1 + kw;
                if (ih >= 0 && ih < H && iw >= 0 && iw < W)
                    m = fmaxf(m, x[(((long long)n * H + ih) * W + iw) * C + c]);
            }
        y[i] = m;
    }
}

// ===================  norms / softmax (wave32 shuffles)  ===================
__global__ void k_layernorm(const float* __restrict__ x, const float* __restrict__ g,
                            const float* __restrict__ b, float* __restrict__ y,
                            long long rows, int C) {
    long long row = (long long)blockIdx.x * (blockDim.x >> 5) + (threadIdx.x >> 5);
    int lane = threadIdx.x & 31;
    if (row >= rows) return;
    const float* p = x + row * C;
    float s = 0.f, s2 = 0.f;
    for (int c = lane; c < C; c += 32) { float v = p[c]; s += v; s2 += v * v; }
    for (int o = 16; o; o >>= 1) { s += __shfl_xor(s, o, 32); s2 += __shfl_xor(s2, o, 32); }
    float mean = s / (float)C;
    float inv  = rsqrtf(s2 / (float)C - mean * mean + 1e-5f);
    float* q = y + row * C;
    for (int c = lane; c < C; c += 32) q[c] = (p[c] - mean) * inv * g[c] + b[c];
}

__global__ void k_softmax(float* __restrict__ x, long long rows, int cols, float scale) {
    long long row = (long long)blockIdx.x * (blockDim.x >> 5) + (threadIdx.x >> 5);
    int lane = threadIdx.x & 31;
    if (row >= rows) return;
    float* p = x + row * cols;
    float mx = -3.4e38f;
    for (int c = lane; c < cols; c += 32) mx = fmaxf(mx, p[c] * scale);
    for (int o = 16; o; o >>= 1) mx = fmaxf(mx, __shfl_xor(mx, o, 32));
    float s = 0.f;
    for (int c = lane; c < cols; c += 32) { float e = __expf(p[c] * scale - mx); p[c] = e; s += e; }
    for (int o = 16; o; o >>= 1) s += __shfl_xor(s, o, 32);
    float inv = 1.0f / s;
    for (int c = lane; c < cols; c += 32) p[c] *= inv;
}

__global__ void k_groupnorm_relu(const float* __restrict__ x, const float* __restrict__ g,
                                 const float* __restrict__ b, float* __restrict__ y,
                                 int HW, int C, int G, int act) {
    int n = blockIdx.x, grp = blockIdx.y;
    int cg = C / G;
    long long cnt = (long long)HW * cg;
    const float* xb = x + (long long)n * HW * C;
    float s = 0.f, s2 = 0.f;
    for (long long i = threadIdx.x; i < cnt; i += blockDim.x) {
        int p = (int)(i / cg); int cc = grp * cg + (int)(i % cg);
        float v = xb[(long long)p * C + cc]; s += v; s2 += v * v;
    }
    __shared__ float sh1[256], sh2[256];
    sh1[threadIdx.x] = s; sh2[threadIdx.x] = s2; __syncthreads();
    for (int o = 128; o; o >>= 1) {
        if ((int)threadIdx.x < o) { sh1[threadIdx.x] += sh1[threadIdx.x + o]; sh2[threadIdx.x] += sh2[threadIdx.x + o]; }
        __syncthreads();
    }
    float mean = sh1[0] / (float)cnt;
    float inv  = rsqrtf(sh2[0] / (float)cnt - mean * mean + 1e-5f);
    float* yb = y + (long long)n * HW * C;
    for (long long i = threadIdx.x; i < cnt; i += blockDim.x) {
        int p = (int)(i / cg); int cc = grp * cg + (int)(i % cg);
        float v = (xb[(long long)p * C + cc] - mean) * inv * g[cc] + b[cc];
        yb[(long long)p * C + cc] = act ? fmaxf(v, 0.f) : v;
    }
}

// ===========================  attention helpers  ===========================
// q,k -> [B*nh, S, hd] bf16 ; v -> transposed [B*nh, hd, Spad] bf16 (pre-zeroed)
__global__ void k_split_heads(const float* __restrict__ qkv, bf16_t* __restrict__ q,
                              bf16_t* __restrict__ k, bf16_t* __restrict__ vT,
                              int Bx, int S, int nh, int hd, int Spad) {
    long long total = (long long)Bx * S * nh * hd;
    int D = nh * hd;
    for (long long i = (long long)blockIdx.x * blockDim.x + threadIdx.x; i < total;
         i += (long long)gridDim.x * blockDim.x) {
        int d = (int)(i % hd); long long t = i / hd;
        int h = (int)(t % nh); t /= nh;
        int s = (int)(t % S);  int b = (int)(t / S);
        const float* base = qkv + (((long long)b * S + s) * 3 * D);
        long long o = (((long long)(b * nh + h)) * S + s) * hd + d;
        q[o] = (bf16_t)base[h * hd + d];
        k[o] = (bf16_t)base[D + h * hd + d];
        vT[(((long long)(b * nh + h)) * hd + d) * Spad + s] = (bf16_t)base[2 * D + h * hd + d];
    }
}

__global__ void k_merge_heads(const float* __restrict__ src, float* __restrict__ dst,
                              int Bx, int S, int nh, int hd) {
    long long total = (long long)Bx * S * nh * hd;
    int D = nh * hd;
    for (long long i = (long long)blockIdx.x * blockDim.x + threadIdx.x; i < total;
         i += (long long)gridDim.x * blockDim.x) {
        int d = (int)(i % hd); long long t = i / hd;
        int h = (int)(t % nh); t /= nh;
        int s = (int)(t % S);  int b = (int)(t / S);
        dst[((long long)b * S + s) * D + h * hd + d] =
            src[(((long long)(b * nh + h)) * S + s) * hd + d];
    }
}

__global__ void k_add_pos(float* __restrict__ t, const float* __restrict__ pos,
                          long long total, long long SD) {
    for (long long i = (long long)blockIdx.x * blockDim.x + threadIdx.x; i < total;
         i += (long long)gridDim.x * blockDim.x)
        t[i] += pos[i % SD];
}

__global__ void k_build_seq(const float* __restrict__ ctx, const float* __restrict__ tok,
                            float* __restrict__ seq, int Bx, int S, int D) {
    long long total = (long long)Bx * (S + 1) * D;
    for (long long i = (long long)blockIdx.x * blockDim.x + threadIdx.x; i < total;
         i += (long long)gridDim.x * blockDim.x) {
        int d = (int)(i % D); long long t = i / D;
        int s = (int)(t % (S + 1)); int b = (int)(t / (S + 1));
        seq[i] = (s == 0) ? ctx[(long long)b * D + d]
                          : tok[((long long)b * S + (s - 1)) * D + d];
    }
}

__global__ void k_drop_first(const float* __restrict__ seq, float* __restrict__ o,
                             int Bx, int S, int D) {
    long long total = (long long)Bx * S * D;
    for (long long i = (long long)blockIdx.x * blockDim.x + threadIdx.x; i < total;
         i += (long long)gridDim.x * blockDim.x) {
        int d = (int)(i % D); long long t = i / D;
        int s = (int)(t % S); int b = (int)(t / S);
        o[i] = seq[((long long)b * (S + 1) + (s + 1)) * D + d];
    }
}

// ===============================  GCN  =====================================
__global__ void k_gcn_deg(const int* __restrict__ dst, float* __restrict__ deg, int E) {
    for (long long i = (long long)blockIdx.x * blockDim.x + threadIdx.x; i < E;
         i += (long long)gridDim.x * blockDim.x)
        atomicAdd(&deg[dst[i]], 1.0f);
}

__global__ void k_rsqrt1(const float* __restrict__ deg, float* __restrict__ dinv, int n) {
    for (long long i = (long long)blockIdx.x * blockDim.x + threadIdx.x; i < n;
         i += (long long)gridDim.x * blockDim.x)
        dinv[i] = rsqrtf(deg[i] + 1.0f);
}

__global__ void k_gcn_scatter(const float* __restrict__ h, const int* __restrict__ src,
                              const int* __restrict__ dst, const float* __restrict__ dinv,
                              float* __restrict__ agg, int E, int C) {
    long long total = (long long)E * C;
    for (long long i = (long long)blockIdx.x * blockDim.x + threadIdx.x; i < total;
         i += (long long)gridDim.x * blockDim.x) {
        int c = (int)(i % C); int e = (int)(i / C);
        int s = src[e], d = dst[e];
        atomicAdd(&agg[(long long)d * C + c], h[(long long)s * C + c] * dinv[s] * dinv[d]);
    }
}

__global__ void k_gcn_self(const float* __restrict__ h, const float* __restrict__ dinv,
                           const float* __restrict__ b, float* __restrict__ agg,
                           long long n, int C) {
    long long total = n * C;
    for (long long i = (long long)blockIdx.x * blockDim.x + threadIdx.x; i < total;
         i += (long long)gridDim.x * blockDim.x) {
        int c = (int)(i % C); long long r = i / C;
        agg[i] += h[i] * dinv[r] * dinv[r] + b[c];
    }
}

__global__ void k_gather_rows(const float* __restrict__ src, const int* __restrict__ idx,
                              float* __restrict__ dst, int nrows, int C) {
    long long total = (long long)nrows * C;
    for (long long i = (long long)blockIdx.x * blockDim.x + threadIdx.x; i < total;
         i += (long long)gridDim.x * blockDim.x) {
        int c = (int)(i % C); int r = (int)(i / C);
        dst[i] = src[(long long)idx[r] * C + c];
    }
}

__global__ void k_copy_cols(float* __restrict__ dst, int Cdst, int coff,
                            const float* __restrict__ src, int Csrc, long long rows) {
    long long total = rows * Csrc;
    for (long long i = (long long)blockIdx.x * blockDim.x + threadIdx.x; i < total;
         i += (long long)gridDim.x * blockDim.x) {
        int c = (int)(i % Csrc); long long r = i / Csrc;
        dst[r * Cdst + coff + c] = src[i];
    }
}

// ===========================  host-side plumbing  ==========================
static inline int ceilDiv(long long a, long long b) { return (int)((a + b - 1) / b); }
static inline dim3 gridFor(long long n) {
    long long b = (n + 255) / 256;
    if (b < 1) b = 1;
    if (b > 262144) b = 262144;
    return dim3((unsigned)b, 1, 1);
}

struct Ctx {
    hipStream_t s;
    char* ws; size_t off, cap;
    const float* prm; long long po;
};
static void* wsAlloc(Ctx& c, size_t bytes) {
    size_t o = (c.off + 255) & ~(size_t)255;
    if (o + bytes > c.cap) o = 0;   // wrap: bounded scratch, no runtime validation here
    c.off = o + bytes;
    return c.ws + o;
}
static const float* take(Ctx& c, long long n) { const float* r = c.prm + c.po; c.po += n; return r; }

struct Affine { float *scale, *shift; };
static Affine bnP(Ctx& c, int C) {
    const float* g = take(c, C); const float* b = take(c, C);
    const float* m = take(c, C); const float* v = take(c, C);
    Affine a;
    a.scale = (float*)wsAlloc(c, (size_t)C * 4);
    a.shift = (float*)wsAlloc(c, (size_t)C * 4);
    k_bn_prep<<<ceilDiv(C, 256), 256, 0, c.s>>>(g, b, m, v, a.scale, a.shift, C);
    return a;
}

static void gemm(hipStream_t s, const bf16_t* A, const bf16_t* B, const float* bias,
                 float* C, int M, int N, int K, int batch,
                 long long sA, long long sB, long long sC, int act) {
    dim3 g(ceilDiv(N, TN), ceilDiv(M, TM), batch);
    k_gemm_wmma<<<g, 256, 0, s>>>(A, B, bias, C, M, N, K, sA, sB, sC, act);
}

// y[rows,N] = act(x[rows,K] * w[N,K]^T + bias); operands bf16-converted, K padded.
static void linear(Ctx& c, const float* x, long long rows, int K, int N,
                   const float* w, const float* bias, int act, float* y) {
    int Kp = align32i(K);
    bf16_t* xb = (bf16_t*)wsAlloc(c, (size_t)(rows * Kp) * 2);
    k_f32_to_bf16_pad<<<gridFor(rows * Kp), 256, 0, c.s>>>(x, xb, rows, K, Kp);
    bf16_t* wb = (bf16_t*)wsAlloc(c, (size_t)N * Kp * 2);
    k_f32_to_bf16_pad<<<gridFor((long long)N * Kp), 256, 0, c.s>>>(w, wb, N, K, Kp);
    gemm(c.s, xb, wb, bias, y, (int)rows, N, Kp, 1, 0, 0, 0, act);
}

// NHWC conv via padded im2col + WMMA GEMM; returns pre-norm NHWC.
static float* convF(Ctx& c, const float* x, int Nb, int H, int W, int Cin,
                    const float* wO, int Cout, int KH, int KW, int stride, int pad,
                    int& OH, int& OW) {
    OH = (H + 2 * pad - KH) / stride + 1;
    OW = (W + 2 * pad - KW) / stride + 1;
    long long rows = (long long)Nb * OH * OW;
    int Kd = Cin * KH * KW;
    int Kp = align32i(Kd);
    bf16_t* col = (bf16_t*)wsAlloc(c, (size_t)(rows * Kp) * 2);
    k_im2col_pad<<<gridFor(rows * Kp), 256, 0, c.s>>>(x, col, Nb, H, W, Cin, KH, KW,
                                                      stride, pad, OH, OW, Kd, Kp);
    bf16_t* wp = (bf16_t*)wsAlloc(c, (size_t)Cout * Kp * 2);
    k_pack_w_pad<<<gridFor((long long)Cout * Kp), 256, 0, c.s>>>(wO, wp, Cout, Cin, KH, KW, Kd, Kp);
    float* y = (float*)wsAlloc(c, (size_t)(rows * Cout) * 4);
    gemm(c.s, col, wp, nullptr, y, (int)rows, Cout, Kp, 1, 0, 0, 0, 0);
    return y;
}

static float* resblock(Ctx& c, float* x, int Nb, int& H, int& W,
                       int Cin, int Cout, int stride, bool ds) {
    const float* w1 = take(c, (long long)Cout * Cin * 9);  Affine bn1 = bnP(c, Cout);
    const float* w2 = take(c, (long long)Cout * Cout * 9); Affine bn2 = bnP(c, Cout);
    const float* dw = nullptr; Affine dbn{nullptr, nullptr};
    if (ds) { dw = take(c, (long long)Cout * Cin); dbn = bnP(c, Cout); }

    int OH, OW;
    float* o1 = convF(c, x, Nb, H, W, Cin, w1, Cout, 3, 3, stride, 1, OH, OW);
    long long rows = (long long)Nb * OH * OW;
    k_affine_act<<<gridFor(rows * Cout), 256, 0, c.s>>>(o1, bn1.scale, bn1.shift, nullptr, o1, rows, Cout, 1);
    int O2, W2;
    float* o2 = convF(c, o1, Nb, OH, OW, Cout, w2, Cout, 3, 3, 1, 1, O2, W2);
    float* idn = x;
    if (ds) {
        int dh, dwv;
        idn = convF(c, x, Nb, H, W, Cin, dw, Cout, 1, 1, stride, 0, dh, dwv);
        k_affine_act<<<gridFor(rows * Cout), 256, 0, c.s>>>(idn, dbn.scale, dbn.shift, nullptr, idn, rows, Cout, 0);
    }
    k_affine_act<<<gridFor(rows * Cout), 256, 0, c.s>>>(o2, bn2.scale, bn2.shift, idn, o2, rows, Cout, 1);
    H = OH; W = OW;
    return o2;
}

struct MhaP { const float *in_w, *in_b, *out_w, *out_b; };
static MhaP mhaP(Ctx& c) {
    MhaP p;
    p.in_w = take(c, 3LL * 128 * 128); p.in_b = take(c, 384);
    p.out_w = take(c, 128LL * 128);    p.out_b = take(c, 128);
    return p;
}

static void mha(Ctx& c, const float* x, int Bx, int S, float* out, const MhaP& p) {
    const int D = 128, nh = 4, hd = 32;
    const int Spad = align32i(S);
    long long rows = (long long)Bx * S;
    float* qkv = (float*)wsAlloc(c, (size_t)(rows * 3 * D) * 4);
    linear(c, x, rows, D, 3 * D, p.in_w, p.in_b, 0, qkv);

    long long hsz = (long long)Bx * nh * S * hd;
    long long vsz = (long long)Bx * nh * hd * Spad;
    bf16_t* qh = (bf16_t*)wsAlloc(c, (size_t)hsz * 2);
    bf16_t* kh = (bf16_t*)wsAlloc(c, (size_t)hsz * 2);
    bf16_t* vT = (bf16_t*)wsAlloc(c, (size_t)vsz * 2);
    hipMemsetAsync(vT, 0, (size_t)vsz * 2, c.s);
    k_split_heads<<<gridFor(hsz), 256, 0, c.s>>>(qkv, qh, kh, vT, Bx, S, nh, hd, Spad);

    long long nsc = (long long)Bx * nh * S * S;
    float* sc = (float*)wsAlloc(c, (size_t)nsc * 4);
    gemm(c.s, qh, kh, nullptr, sc, S, S, hd, Bx * nh,
         (long long)S * hd, (long long)S * hd, (long long)S * S, 0);
    long long srows = (long long)Bx * nh * S;
    k_softmax<<<ceilDiv(srows, 8), 256, 0, c.s>>>(sc, srows, S, 0.1767766953f);

    bf16_t* scb = (bf16_t*)wsAlloc(c, (size_t)(srows * Spad) * 2);
    k_f32_to_bf16_pad<<<gridFor(srows * Spad), 256, 0, c.s>>>(sc, scb, srows, S, Spad);

    float* oh = (float*)wsAlloc(c, (size_t)hsz * 4);
    gemm(c.s, scb, vT, nullptr, oh, S, hd, Spad, Bx * nh,
         (long long)S * Spad, (long long)hd * Spad, (long long)S * hd, 0);

    float* om = (float*)wsAlloc(c, (size_t)(rows * D) * 4);
    k_merge_heads<<<gridFor(hsz), 256, 0, c.s>>>(oh, om, Bx, S, nh, hd);
    linear(c, om, rows, D, D, p.out_w, p.out_b, 0, out);
}

static void txLayer(Ctx& c, float* tok, int Bx, int S) {
    MhaP ap = mhaP(c);
    const float* ln1g = take(c, 128); const float* ln1b = take(c, 128);
    const float* w1 = take(c, 512LL * 128); const float* b1 = take(c, 512);
    const float* w2 = take(c, 128LL * 512); const float* b2 = take(c, 128);
    const float* ln2g = take(c, 128); const float* ln2b = take(c, 128);
    long long rows = (long long)Bx * S;
    float* m = (float*)wsAlloc(c, (size_t)(rows * 128) * 4);
    mha(c, tok, Bx, S, m, ap);
    k_affine_act<<<gridFor(rows * 128), 256, 0, c.s>>>(m, nullptr, nullptr, tok, m, rows, 128, 0);
    k_layernorm<<<ceilDiv(rows, 8), 256, 0, c.s>>>(m, ln1g, ln1b, tok, rows, 128);
    float* f1 = (float*)wsAlloc(c, (size_t)(rows * 512) * 4);
    linear(c, tok, rows, 128, 512, w1, b1, 1, f1);
    float* f2 = (float*)wsAlloc(c, (size_t)(rows * 128) * 4);
    linear(c, f1, rows, 512, 128, w2, b2, 0, f2);
    k_affine_act<<<gridFor(rows * 128), 256, 0, c.s>>>(f2, nullptr, nullptr, tok, f2, rows, 128, 0);
    k_layernorm<<<ceilDiv(rows, 8), 256, 0, c.s>>>(f2, ln2g, ln2b, tok, rows, 128);
}

static float* upT(Ctx& c, const float* x, int Nb, int H, int W, int Cin, int Cout,
                  const float* w, const float* bias, const float* gg, const float* gb) {
    long long rows = (long long)Nb * H * W;
    int Cinp = align32i(Cin);
    bf16_t* xb = (bf16_t*)wsAlloc(c, (size_t)(rows * Cinp) * 2);
    k_f32_to_bf16_pad<<<gridFor(rows * Cinp), 256, 0, c.s>>>(x, xb, rows, Cin, Cinp);
    bf16_t* wp = (bf16_t*)wsAlloc(c, (size_t)Cout * 4 * Cinp * 2);
    k_pack_wt_pad<<<gridFor((long long)Cout * 4 * Cinp), 256, 0, c.s>>>(w, wp, Cin, Cout, Cinp);
    float* yg = (float*)wsAlloc(c, (size_t)(rows * Cout * 4) * 4);
    gemm(c.s, xb, wp, nullptr, yg, (int)rows, Cout * 4, Cinp, 1, 0, 0, 0, 0);
    float* out = (float*)wsAlloc(c, (size_t)((long long)Nb * 4 * H * W * Cout) * 4);
    k_convt_scatter<<<gridFor(rows * Cout * 4), 256, 0, c.s>>>(yg, bias, out, Nb, H, W, Cout);
    dim3 gng(Nb, 8);
    k_groupnorm_relu<<<gng, 256, 0, c.s>>>(out, gg, gb, out, 4 * H * W, Cout, 8, 1);
    return out;
}

static float* cvB(Ctx& c, const float* x, int Nb, int H, int W, int Cin, int Cout,
                  const float* w, const float* bias, const float* gg, const float* gb) {
    int OH, OW;
    float* y = convF(c, x, Nb, H, W, Cin, w, Cout, 3, 3, 1, 1, OH, OW);
    long long rows = (long long)Nb * H * W;
    k_affine_act<<<gridFor(rows * Cout), 256, 0, c.s>>>(y, nullptr, bias, nullptr, y, rows, Cout, 0);
    dim3 gng(Nb, 8);
    k_groupnorm_relu<<<gng, 256, 0, c.s>>>(y, gg, gb, y, H * W, Cout, 8, 1);
    return y;
}

// ===========================  kernel_launch  ===============================
extern "C" void kernel_launch(void* const* d_in, const int* in_sizes, int n_in,
                              void* d_out, int out_size, void* d_ws, size_t ws_size,
                              hipStream_t stream) {
    (void)in_sizes; (void)n_in; (void)out_size;
    const float* patches = (const float*)d_in[0];   // [8,3,256,256]
    const float* nodef   = (const float*)d_in[1];   // [50000,1536]
    const int*   edges   = (const int*)d_in[2];     // [2,800000]
    const int*   tgt     = (const int*)d_in[3];     // [8]
    const float* prm     = (const float*)d_in[4];   // flattened params (insertion order)
    float* outp = (float*)d_out;

    const int Bb = 8, E = 800000, NN = 50000;
    Ctx c{stream, (char*)d_ws, 0, ws_size, prm, 0};

    // ---- CNN encoder (NHWC) ----
    float* xh = (float*)wsAlloc(c, (size_t)(8LL * 256 * 256 * 3) * 4);
    k_nchw2nhwc<<<gridFor(8LL * 3 * 256 * 256), 256, 0, stream>>>(patches, xh, Bb, 3, 256, 256);

    const float* stem_w = take(c, 64LL * 3 * 49);
    Affine sbn = bnP(c, 64);
    int OH, OW;
    float* s0 = convF(c, xh, Bb, 256, 256, 3, stem_w, 64, 7, 7, 2, 3, OH, OW);   // 128x128x64
    k_affine_act<<<gridFor(8LL * 128 * 128 * 64), 256, 0, stream>>>(
        s0, sbn.scale, sbn.shift, nullptr, s0, 8LL * 128 * 128, 64, 1);
    float* p0 = (float*)wsAlloc(c, (size_t)(8LL * 64 * 64 * 64) * 4);
    k_maxpool<<<gridFor(8LL * 64 * 64 * 64), 256, 0, stream>>>(s0, p0, Bb, 128, 128, 64, 64, 64);

    int H = 64, W = 64;
    float* t = p0;
    t = resblock(c, t, Bb, H, W, 64, 64, 1, false);
    t = resblock(c, t, Bb, H, W, 64, 64, 1, false);
    float* f64 = t;                                                  // 64x64x64
    t = resblock(c, t, Bb, H, W, 64, 128, 2, true);
    t = resblock(c, t, Bb, H, W, 128, 128, 1, false);
    float* f32p = t;                                                 // 32x32x128
    t = resblock(c, t, Bb, H, W, 128, 256, 2, true);
    t = resblock(c, t, Bb, H, W, 256, 256, 1, false);                // 16x16x256

    // proj 1x1 + positional embedding
    const float* proj_w = take(c, 128LL * 256);
    const float* proj_b = take(c, 128);
    float* tok = (float*)wsAlloc(c, (size_t)(8LL * 256 * 128) * 4);
    linear(c, t, 8LL * 256, 256, 128, proj_w, proj_b, 0, tok);
    const float* pos = take(c, 256LL * 128);
    k_add_pos<<<gridFor(8LL * 256 * 128), 256, 0, stream>>>(tok, pos, 8LL * 256 * 128, 256LL * 128);

    for (int l = 0; l < 2; ++l) txLayer(c, tok, Bb, 256);

    // ---- GCN ----
    const float* gpw = take(c, 128LL * 1536); const float* gpb = take(c, 128);
    const float* gl1g = take(c, 128); const float* gl1b = take(c, 128);
    float* h0 = (float*)wsAlloc(c, (size_t)((long long)NN * 128) * 4);
    linear(c, nodef, NN, 1536, 128, gpw, gpb, 0, h0);
    k_layernorm<<<ceilDiv((long long)NN, 8), 256, 0, stream>>>(h0, gl1g, gl1b, h0, NN, 128);

    float* deg = (float*)wsAlloc(c, (size_t)NN * 4);
    hipMemsetAsync(deg, 0, (size_t)NN * 4, stream);
    k_gcn_deg<<<gridFor(E), 256, 0, stream>>>(edges + E, deg, E);
    float* dinv = (float*)wsAlloc(c, (size_t)NN * 4);
    k_rsqrt1<<<gridFor(NN), 256, 0, stream>>>(deg, dinv, NN);

    float* hs = (float*)wsAlloc(c, (size_t)((long long)NN * 512) * 4);
    k_copy_cols<<<gridFor((long long)NN * 128), 256, 0, stream>>>(hs, 512, 0, h0, 128, NN);
    float* hcur = h0;
    for (int l = 0; l < 3; ++l) {
        const float* w = take(c, 128LL * 128);
        const float* b = take(c, 128);
        float* hw = (float*)wsAlloc(c, (size_t)((long long)NN * 128) * 4);
        linear(c, hcur, NN, 128, 128, w, nullptr, 0, hw);
        float* agg = (float*)wsAlloc(c, (size_t)((long long)NN * 128) * 4);
        hipMemsetAsync(agg, 0, (size_t)((long long)NN * 128) * 4, stream);
        k_gcn_scatter<<<gridFor((long long)E * 128), 256, 0, stream>>>(hw, edges, edges + E, dinv, agg, E, 128);
        k_gcn_self<<<gridFor((long long)NN * 128), 256, 0, stream>>>(hw, dinv, b, agg, NN, 128);
        k_copy_cols<<<gridFor((long long)NN * 128), 256, 0, stream>>>(hs, 512, (l + 1) * 128, agg, 128, NN);
        hcur = agg;
    }
    const float* gmw = take(c, 128LL * 512); const float* gmb = take(c, 128);
    const float* gl2g = take(c, 128); const float* gl2b = take(c, 128);
    float* ctxall = (float*)wsAlloc(c, (size_t)((long long)NN * 128) * 4);
    linear(c, hs, NN, 512, 128, gmw, gmb, 0, ctxall);
    k_layernorm<<<ceilDiv((long long)NN, 8), 256, 0, stream>>>(ctxall, gl2g, gl2b, ctxall, NN, 128);
    float* ctxv = (float*)wsAlloc(c, (size_t)(8 * 128) * 4);
    k_gather_rows<<<gridFor(8 * 128), 256, 0, stream>>>(ctxall, tgt, ctxv, 8, 128);

    // ---- fusion cross-attention (S = 257) ----
    const float* lng = take(c, 128); const float* lnb = take(c, 128);
    MhaP fap = mhaP(c);
    long long fr = 8LL * 257;
    float* seq = (float*)wsAlloc(c, (size_t)(fr * 128) * 4);
    k_build_seq<<<gridFor(fr * 128), 256, 0, stream>>>(ctxv, tok, seq, Bb, 256, 128);
    float* lnseq = (float*)wsAlloc(c, (size_t)(fr * 128) * 4);
    k_layernorm<<<ceilDiv(fr, 8), 256, 0, stream>>>(seq, lng, lnb, lnseq, fr, 128);
    float* mo = (float*)wsAlloc(c, (size_t)(fr * 128) * 4);
    mha(c, lnseq, Bb, 257, mo, fap);
    k_affine_act<<<gridFor(fr * 128), 256, 0, stream>>>(mo, nullptr, nullptr, seq, seq, fr, 128, 0);
    float* feat = (float*)wsAlloc(c, (size_t)(8LL * 256 * 128) * 4);
    k_drop_first<<<gridFor(8LL * 256 * 128), 256, 0, stream>>>(seq, feat, Bb, 256, 128);

    // ---- decoder (NHWC) ----
    const float* u1w = take(c, 128LL * 128 * 4); const float* u1b = take(c, 128);
    const float* u1g = take(c, 128); const float* u1nb = take(c, 128);
    float* d1 = upT(c, feat, Bb, 16, 16, 128, 128, u1w, u1b, u1g, u1nb);        // 32x32x128
    const float* c1w = take(c, 128LL * 256 * 9); const float* c1b = take(c, 128);
    const float* c1g = take(c, 128); const float* c1nb = take(c, 128);
    float* cat1 = (float*)wsAlloc(c, (size_t)(8LL * 32 * 32 * 256) * 4);
    k_copy_cols<<<gridFor(8LL * 1024 * 128), 256, 0, stream>>>(cat1, 256, 0, d1, 128, 8LL * 1024);
    k_copy_cols<<<gridFor(8LL * 1024 * 128), 256, 0, stream>>>(cat1, 256, 128, f32p, 128, 8LL * 1024);
    float* e1 = cvB(c, cat1, Bb, 32, 32, 256, 128, c1w, c1b, c1g, c1nb);

    const float* u2w = take(c, 128LL * 64 * 4); const float* u2b = take(c, 64);
    const float* u2g = take(c, 64); const float* u2nb = take(c, 64);
    float* d2 = upT(c, e1, Bb, 32, 32, 128, 64, u2w, u2b, u2g, u2nb);           // 64x64x64
    const float* c2w = take(c, 64LL * 128 * 9); const float* c2b = take(c, 64);
    const float* c2g = take(c, 64); const float* c2nb = take(c, 64);
    float* cat2 = (float*)wsAlloc(c, (size_t)(8LL * 64 * 64 * 128) * 4);
    k_copy_cols<<<gridFor(8LL * 4096 * 64), 256, 0, stream>>>(cat2, 128, 0, d2, 64, 8LL * 4096);
    k_copy_cols<<<gridFor(8LL * 4096 * 64), 256, 0, stream>>>(cat2, 128, 64, f64, 64, 8LL * 4096);
    float* e2 = cvB(c, cat2, Bb, 64, 64, 128, 64, c2w, c2b, c2g, c2nb);

    const float* u3w = take(c, 64LL * 32 * 4); const float* u3b = take(c, 32);
    const float* u3g = take(c, 32); const float* u3nb = take(c, 32);
    float* d3 = upT(c, e2, Bb, 64, 64, 64, 32, u3w, u3b, u3g, u3nb);            // 128x128x32
    const float* c3w = take(c, 32LL * 32 * 9); const float* c3b = take(c, 32);
    const float* c3g = take(c, 32); const float* c3nb = take(c, 32);
    float* e3 = cvB(c, d3, Bb, 128, 128, 32, 32, c3w, c3b, c3g, c3nb);

    const float* u4w = take(c, 32LL * 16 * 4); const float* u4b = take(c, 16);
    const float* u4g = take(c, 16); const float* u4nb = take(c, 16);
    float* d4 = upT(c, e3, Bb, 128, 128, 32, 16, u4w, u4b, u4g, u4nb);          // 256x256x16
    const float* c4w = take(c, 16LL * 16 * 9); const float* c4b = take(c, 16);
    const float* c4g = take(c, 16); const float* c4nb = take(c, 16);
    float* e4 = cvB(c, d4, Bb, 256, 256, 16, 16, c4w, c4b, c4g, c4nb);

    // head 1x1 -> NCHW output
    const float* hw_ = take(c, 3LL * 16); const float* hb_ = take(c, 3);
    float* logits = (float*)wsAlloc(c, (size_t)(8LL * 256 * 256 * 3) * 4);
    linear(c, e4, 8LL * 256 * 256, 16, 3, hw_, hb_, 0, logits);
    k_nhwc2nchw<<<gridFor(8LL * 256 * 256 * 3), 256, 0, stream>>>(logits, outp, Bb, 256, 256, 3);
}
